// MaskedCausalAttention_86517821215604
// MI455X (gfx1250) — matrix operand order repaired and checked by reading, not compiled, not run on hardware
//
#include <hip/hip_runtime.h>
#include <hip/hip_bf16.h>
#include <stdint.h>

typedef __attribute__((ext_vector_type(16))) __bf16 v16bf;
typedef __attribute__((ext_vector_type(8)))  __bf16 v8bf;
typedef __attribute__((ext_vector_type(8)))  float  v8f;

#define TDIM 2048
#define CDIM 1024
#define NH   16
#define DH   64
#define BB   4
#define QT_PER_WG 4   // 4 waves x 16 queries = 64 queries per workgroup

__device__ __forceinline__ v8f wmma_bf16(v16bf a, v16bf b, v8f c) {
  // D = A(16x32 bf16) * B(32x16 bf16) + C(16x16 f32)
  return __builtin_amdgcn_wmma_f32_16x16x32_bf16(false, a, false, b, (short)0, c,
                                                 false, false);
}

// A-fragment (16x32, row-major source, row stride ld elements).
// lanes 0-15: row = lane, K = {0..7, 16..23}; lanes 16-31: row = lane-16, K = {8..15, 24..31}
__device__ __forceinline__ v16bf load_frag_a(const __bf16* p, int ld) {
  int lane = threadIdx.x & 31;
  int row  = lane & 15;
  int koff = (lane >> 4) << 3;  // 0 or 8
  const __bf16* r = p + (size_t)row * ld + koff;
  v8bf lo = *(const v8bf*)(r);
  v8bf hi = *(const v8bf*)(r + 16);
  v16bf out;
#pragma unroll
  for (int i = 0; i < 8; ++i) { out[i] = lo[i]; out[i + 8] = hi[i]; }
  return out;
}

// B-fragment (32x16) loaded from B^T stored row-major [N][K], row stride ld.
// lanes 0-15: col = lane, K = {0..7, 16..23}; lanes 16-31: col = lane-16, K = {8..15, 24..31}
__device__ __forceinline__ v16bf load_frag_b(const __bf16* p, int ld) {
  int lane = threadIdx.x & 31;
  int col  = lane & 15;
  int koff = (lane >> 4) << 3;  // 0 or 8
  const __bf16* r = p + (size_t)col * ld + koff;
  v8bf lo = *(const v8bf*)(r);
  v8bf hi = *(const v8bf*)(r + 16);
  v16bf out;
#pragma unroll
  for (int i = 0; i < 8; ++i) { out[i] = lo[i]; out[i + 8] = hi[i]; }
  return out;
}

// Async copy of 16 bytes global -> LDS (CDNA5 GLOBAL_LOAD_ASYNC_TO_LDS_B128, GV mode).
// lds_off: LDS byte offset; tracked by ASYNCcnt.
__device__ __forceinline__ void async_copy16(unsigned lds_off, const __bf16* g) {
  asm volatile("global_load_async_to_lds_b128 %0, %1, off"
               :: "v"(lds_off), "v"((unsigned long long)(uintptr_t)g)
               : "memory");
}

__device__ __forceinline__ void wait_async0() {
  asm volatile("s_wait_asynccnt 0" ::: "memory");
}

__global__ void cvt_f32_bf16(const float* __restrict__ in, __bf16* __restrict__ out, int n) {
  int i = blockIdx.x * 256 + threadIdx.x;
  if (i < n) out[i] = (__bf16)in[i];
}

// wt[n][k] = (bf16) w[k][n]   (C x C)
__global__ void cvt_wt(const float* __restrict__ w, __bf16* __restrict__ wt) {
  int k = blockIdx.x * 16 + threadIdx.x;
  int n = blockIdx.y * 16 + threadIdx.y;
  wt[(size_t)n * CDIM + k] = (__bf16)w[(size_t)k * CDIM + n];
}

// Fused Q/K/V projection. One wave -> 32x16 tile of each of Q, K, V.
// 2 A-frags + 3 B-frags -> 6 WMMAs per k-step.
// Q,K stored bf16 [B*T, C]; V stored transposed per head: Vt[b][h][d][t] bf16.
__global__ void qkv_gemm(const __bf16* __restrict__ xb,
                         const __bf16* __restrict__ wqt, const __bf16* __restrict__ wkt,
                         const __bf16* __restrict__ wvt,
                         const float* __restrict__ bq, const float* __restrict__ bk,
                         const float* __restrict__ bv,
                         __bf16* __restrict__ Qb, __bf16* __restrict__ Kb,
                         __bf16* __restrict__ Vt) {
  const int row0 = blockIdx.x * 32;
  const int n0   = blockIdx.y * 16;
  v8f cq0 = {}, cq1 = {}, ck0 = {}, ck1 = {}, cv0 = {}, cv1 = {};
  const __bf16* arow0 = xb + (size_t)row0 * CDIM;
  const __bf16* arow1 = xb + (size_t)(row0 + 16) * CDIM;
  for (int k = 0; k < CDIM; k += 32) {
    v16bf a0 = load_frag_a(arow0 + k, CDIM);
    v16bf a1 = load_frag_a(arow1 + k, CDIM);
    if (k + 32 < CDIM) {  // prefetch next k-step of the streaming operand
      __builtin_prefetch(arow0 + k + 32, 0, 1);
      __builtin_prefetch(arow1 + k + 32, 0, 1);
    }
    v16bf bqf = load_frag_b(wqt + (size_t)n0 * CDIM + k, CDIM);
    cq0 = wmma_bf16(a0, bqf, cq0);
    cq1 = wmma_bf16(a1, bqf, cq1);
    v16bf bkf = load_frag_b(wkt + (size_t)n0 * CDIM + k, CDIM);
    ck0 = wmma_bf16(a0, bkf, ck0);
    ck1 = wmma_bf16(a1, bkf, ck1);
    v16bf bvf = load_frag_b(wvt + (size_t)n0 * CDIM + k, CDIM);
    cv0 = wmma_bf16(a0, bvf, cv0);
    cv1 = wmma_bf16(a1, bvf, cv1);
  }
  const int lane  = threadIdx.x & 31;
  const int col   = lane & 15;
  const int rbase = (lane >> 4) * 8;
  const float vbq = bq[n0 + col], vbk = bk[n0 + col], vbv = bv[n0 + col];
  const int cn = n0 + col;
  const int h  = cn >> 6;          // / DH
  const int d  = cn & (DH - 1);
#pragma unroll
  for (int half = 0; half < 2; ++half) {
    v8f cq = half ? cq1 : cq0;
    v8f ck = half ? ck1 : ck0;
    v8f cv = half ? cv1 : cv0;
#pragma unroll
    for (int r = 0; r < 8; ++r) {
      int row = row0 + half * 16 + rbase + r;
      size_t idx = (size_t)row * CDIM + cn;
      Qb[idx] = (__bf16)(cq[r] + vbq);
      Kb[idx] = (__bf16)(ck[r] + vbk);
      int bidx = row >> 11;        // / TDIM
      int t    = row & (TDIM - 1);
      Vt[(((size_t)bidx * NH + h) * DH + d) * TDIM + t] = (__bf16)(cv[r] + vbv);
    }
  }
}

// Flash attention: 4-wave workgroup per (b, h, 64-query block).
// K/V chunks staged once per workgroup into LDS via async-to-LDS, shared by all 4 waves.
__global__ void attn_fwd(const __bf16* __restrict__ Qb, const __bf16* __restrict__ Kb,
                         const __bf16* __restrict__ Vt, __bf16* __restrict__ Ob) {
  const int q0_blk = blockIdx.x * (16 * QT_PER_WG);
  const int h  = blockIdx.y;
  const int b  = blockIdx.z;
  const int tid  = threadIdx.x;        // 0..127
  const int w    = tid >> 5;           // wave id 0..3
  const int lane = tid & 31;
  const int col  = lane & 15;
  const int rbase = (lane >> 4) * 8;
  const int q0 = q0_blk + w * 16;

  __shared__ alignas(16) __bf16 kshm[32 * DH];            // [key][d]  (32 x 64)
  __shared__ alignas(16) __bf16 vshm[DH * 32];            // [d][t]    (64 x 32)
  __shared__ alignas(16) __bf16 pshm[QT_PER_WG][16 * 32]; // per-wave P tile

  const __bf16* qbase = Qb + (((size_t)b * TDIM + q0) * CDIM) + h * DH;
  v16bf qa0 = load_frag_a(qbase, CDIM);       // d = 0..31
  v16bf qa1 = load_frag_a(qbase + 32, CDIM);  // d = 32..63

  v8f o0 = {}, o1 = {}, o2 = {}, o3 = {};
  float m[8], l[8];
#pragma unroll
  for (int r = 0; r < 8; ++r) { m[r] = -1e30f; l[r] = 0.f; }
  const float scale = 0.125f;  // 1/sqrt(64)
  const int kend_w   = q0 + 16;                 // keys this wave needs (causal)
  const int kend_blk = q0_blk + 16 * QT_PER_WG; // keys the block needs

  const unsigned klds = (unsigned)(uintptr_t)&kshm[0];
  const unsigned vlds = (unsigned)(uintptr_t)&vshm[0];
  const __bf16* kgbase = Kb + ((size_t)b * TDIM) * CDIM + h * DH;
  const __bf16* vgbase = Vt + (((size_t)b * NH + h) * DH) * TDIM;

  for (int s0 = 0; s0 < kend_blk; s0 += 32) {
    // ---- cooperative async stage: K chunk 32x64 (4KB), V chunk 64x32 (4KB)
#pragma unroll
    for (int it = 0; it < 2; ++it) {
      int e = tid + it * 128;  // 256 x 16B chunks each
      // K: row = key (8 chunks per row of 64 bf16)
      const __bf16* kg = kgbase + (size_t)(s0 + (e >> 3)) * CDIM + (e & 7) * 8;
      async_copy16(klds + e * 16, kg);
      // V: row = d (4 chunks per row of 32 bf16)
      const __bf16* vg = vgbase + (size_t)(e >> 2) * TDIM + s0 + (e & 3) * 8;
      async_copy16(vlds + e * 16, vg);
    }
    wait_async0();
    __syncthreads();   // uniform: all waves reach here

    if (s0 < kend_w) {  // wave-uniform predicate; EXEC stays full for WMMA
      // ---- scores from LDS-resident K
      v8f sc0 = {}, sc1 = {};
      {
        v16bf k0 = load_frag_b(kshm, DH);
        v16bf k1 = load_frag_b(kshm + 32, DH);
        sc0 = wmma_bf16(qa0, k0, sc0);
        sc0 = wmma_bf16(qa1, k1, sc0);
      }
      const int  s1  = s0 + 16;
      const bool t1v = (s1 < kend_w);
      if (t1v) {
        v16bf k0 = load_frag_b(kshm + 16 * DH, DH);
        v16bf k1 = load_frag_b(kshm + 16 * DH + 32, DH);
        sc1 = wmma_bf16(qa0, k0, sc1);
        sc1 = wmma_bf16(qa1, k1, sc1);
      }
      // ---- online softmax (row r lives in element r across a 16-lane half)
      __bf16* pw = &pshm[w][0];
#pragma unroll
      for (int r = 0; r < 8; ++r) {
        const int qrow = q0 + rbase + r;
        float x0 = sc0[r] * scale;
        if (s0 + col > qrow) x0 = -1e30f;
        float x1 = (t1v && (s1 + col <= qrow)) ? sc1[r] * scale : -1e30f;
        float cm = fmaxf(x0, x1);
        cm = fmaxf(cm, __shfl_xor(cm, 1, 32));
        cm = fmaxf(cm, __shfl_xor(cm, 2, 32));
        cm = fmaxf(cm, __shfl_xor(cm, 4, 32));
        cm = fmaxf(cm, __shfl_xor(cm, 8, 32));
        float mn    = fmaxf(m[r], cm);
        float alpha = __expf(m[r] - mn);
        float p0    = __expf(x0 - mn);
        float p1    = __expf(x1 - mn);
        float rs = p0 + p1;
        rs += __shfl_xor(rs, 1, 32);
        rs += __shfl_xor(rs, 2, 32);
        rs += __shfl_xor(rs, 4, 32);
        rs += __shfl_xor(rs, 8, 32);
        l[r] = l[r] * alpha + rs;
        m[r] = mn;
        o0[r] *= alpha; o1[r] *= alpha; o2[r] *= alpha; o3[r] *= alpha;
        const int prow = rbase + r;
        pw[prow * 32 + col]      = (__bf16)p0;
        pw[prow * 32 + col + 16] = (__bf16)p1;
      }
      // same-wave LDS ops are in-order; fence only the compiler's scheduling
      __builtin_amdgcn_wave_barrier();
      v16bf pa = load_frag_a(pw, 32);  // P as A-fragment (16x32)
      __builtin_amdgcn_wave_barrier();
      // ---- O += P * V from LDS-resident V (4 d-tiles of 16 columns)
      o0 = wmma_bf16(pa, load_frag_b(vshm,           32), o0);
      o1 = wmma_bf16(pa, load_frag_b(vshm + 16 * 32, 32), o1);
      o2 = wmma_bf16(pa, load_frag_b(vshm + 32 * 32, 32), o2);
      o3 = wmma_bf16(pa, load_frag_b(vshm + 48 * 32, 32), o3);
    }
    __syncthreads();   // uniform: protect K/V LDS before next stage
  }

  __bf16* obase = Ob + (((size_t)b * TDIM + q0) * CDIM) + h * DH;
#pragma unroll
  for (int r = 0; r < 8; ++r) {
    float inv = 1.0f / l[r];
    __bf16* dst = obase + (size_t)(rbase + r) * CDIM;
    dst[col]      = (__bf16)(o0[r] * inv);
    dst[16 + col] = (__bf16)(o1[r] * inv);
    dst[32 + col] = (__bf16)(o2[r] * inv);
    dst[48 + col] = (__bf16)(o3[r] * inv);
  }
}

// out = Ob @ Wo + bo  (fp32 output). One wave -> 32x64 tile: 2 A + 4 B -> 8 WMMAs/k-step.
__global__ void out_proj(const __bf16* __restrict__ Ob, const __bf16* __restrict__ wot,
                         const float* __restrict__ bo, float* __restrict__ out) {
  const int row0 = blockIdx.x * 32;
  const int n0   = blockIdx.y * 64;
  v8f c0[4] = {}, c1[4] = {};
  const __bf16* arow0 = Ob + (size_t)row0 * CDIM;
  const __bf16* arow1 = Ob + (size_t)(row0 + 16) * CDIM;
  for (int k = 0; k < CDIM; k += 32) {
    v16bf a0 = load_frag_a(arow0 + k, CDIM);
    v16bf a1 = load_frag_a(arow1 + k, CDIM);
    if (k + 32 < CDIM) {
      __builtin_prefetch(arow0 + k + 32, 0, 1);
      __builtin_prefetch(arow1 + k + 32, 0, 1);
    }
#pragma unroll
    for (int j = 0; j < 4; ++j) {
      v16bf wfr = load_frag_b(wot + (size_t)(n0 + j * 16) * CDIM + k, CDIM);
      c0[j] = wmma_bf16(a0, wfr, c0[j]);
      c1[j] = wmma_bf16(a1, wfr, c1[j]);
    }
  }
  const int col   = threadIdx.x & 15;
  const int rbase = ((threadIdx.x & 31) >> 4) * 8;
#pragma unroll
  for (int j = 0; j < 4; ++j) {
    const float bias = bo[n0 + j * 16 + col];
#pragma unroll
    for (int r = 0; r < 8; ++r) {
      out[(size_t)(row0 + rbase + r) * CDIM + n0 + j * 16 + col] = c0[j][r] + bias;
      out[(size_t)(row0 + 16 + rbase + r) * CDIM + n0 + j * 16 + col] = c1[j][r] + bias;
    }
  }
}

extern "C" void kernel_launch(void* const* d_in, const int* in_sizes, int n_in,
                              void* d_out, int out_size, void* d_ws, size_t ws_size,
                              hipStream_t stream) {
  (void)in_sizes; (void)n_in; (void)out_size; (void)ws_size;
  const float* x  = (const float*)d_in[0];
  const float* Wq = (const float*)d_in[1];
  const float* bq = (const float*)d_in[2];
  const float* Wk = (const float*)d_in[3];
  const float* bk = (const float*)d_in[4];
  const float* Wv = (const float*)d_in[5];
  const float* bv = (const float*)d_in[6];
  const float* Wo = (const float*)d_in[7];
  const float* bo = (const float*)d_in[8];
  float* out = (float*)d_out;

  const size_t ROWS = (size_t)BB * TDIM;   // 8192
  const size_t XSZ  = ROWS * CDIM;         // 8M elements
  const size_t WSZ  = (size_t)CDIM * CDIM; // 1M elements

  char* ws = (char*)d_ws;
  size_t off = 0;
  __bf16* xb  = (__bf16*)(ws + off); off += XSZ * sizeof(__bf16);
  __bf16* Qb  = (__bf16*)(ws + off); off += XSZ * sizeof(__bf16);
  __bf16* Kb  = (__bf16*)(ws + off); off += XSZ * sizeof(__bf16);
  __bf16* Vt  = (__bf16*)(ws + off); off += XSZ * sizeof(__bf16);
  __bf16* Ob  = (__bf16*)(ws + off); off += XSZ * sizeof(__bf16);
  __bf16* Wqt = (__bf16*)(ws + off); off += WSZ * sizeof(__bf16);
  __bf16* Wkt = (__bf16*)(ws + off); off += WSZ * sizeof(__bf16);
  __bf16* Wvt = (__bf16*)(ws + off); off += WSZ * sizeof(__bf16);
  __bf16* Wot = (__bf16*)(ws + off); off += WSZ * sizeof(__bf16);

  cvt_f32_bf16<<<dim3((unsigned)(XSZ / 256)), dim3(256), 0, stream>>>(x, xb, (int)XSZ);
  dim3 wtg(CDIM / 16, CDIM / 16), wtb(16, 16);
  cvt_wt<<<wtg, wtb, 0, stream>>>(Wq, Wqt);
  cvt_wt<<<wtg, wtb, 0, stream>>>(Wk, Wkt);
  cvt_wt<<<wtg, wtb, 0, stream>>>(Wv, Wvt);
  cvt_wt<<<wtg, wtb, 0, stream>>>(Wo, Wot);

  qkv_gemm<<<dim3((unsigned)(ROWS / 32), CDIM / 16), dim3(32), 0, stream>>>(
      xb, Wqt, Wkt, Wvt, bq, bk, bv, Qb, Kb, Vt);
  attn_fwd<<<dim3(TDIM / (16 * QT_PER_WG), NH, BB), dim3(32 * QT_PER_WG), 0, stream>>>(
      Qb, Kb, Vt, Ob);
  out_proj<<<dim3((unsigned)(ROWS / 32), CDIM / 64), dim3(32), 0, stream>>>(
      Ob, Wot, bo, out);
}
